// GINEFiLMBlock_53077205844804
// MI455X (gfx1250) — matrix operand order repaired and checked by reading, not compiled, not run on hardware
//
#include <hip/hip_runtime.h>
#include <hip/hip_bf16.h>
#include <math.h>

typedef __attribute__((ext_vector_type(16))) _Float16 v16h;
typedef __attribute__((ext_vector_type(8)))  _Float16 v8h;
typedef __attribute__((ext_vector_type(8)))  float    v8f;
typedef __attribute__((ext_vector_type(4)))  float    v4f;
typedef __attribute__((ext_vector_type(4)))  int      v4i;

#define NN 100000
#define EE 1600000
#define HH 128
#define EDD 32
#define TDD 256
#define GG 512
#define EPS_GN 1e-5f

// ---------------------------------------------------------------------------
// Prep: convert a [K x 128] f32 weight matrix into f16 B-operand fragments.
//   frag f = kt*8+nt, slot = f*32+lane:
//   out[slot*16 + j] = W[kt*32 + (lane<16 ? 0 : 8) + j + (j<8 ? 0 : 8)][nt*16 + (lane&15)]
// ---------------------------------------------------------------------------
__global__ void prep_frag_kernel(const float* __restrict__ W,
                                 _Float16* __restrict__ out, int KT) {
  const int total = KT * 8 * 32;
  for (int s = blockIdx.x * blockDim.x + threadIdx.x; s < total;
       s += gridDim.x * blockDim.x) {
    const int f = s >> 5, ln = s & 31;
    const int kt = f >> 3, nt = f & 7;
    const int n  = nt * 16 + (ln & 15);
    const int kb = kt * 32 + ((ln < 16) ? 0 : 8);
    #pragma unroll
    for (int j = 0; j < 16; ++j)
      out[s * 16 + j] = (_Float16)W[(kb + j + (j < 8 ? 0 : 8)) * HH + n];
  }
}

// ---------------------------------------------------------------------------
// Kernel 1: fused edge stage.
//   msg = relu(x[src] + edge_attr @ We + be);  agg[dst] += msg
// 32 edges per wave. Streamed inputs (edge_attr, indices) use non-temporal
// loads so x/agg stay L2-resident. Gathers are batched per tile before the
// atomic loop so loads overlap.
// ---------------------------------------------------------------------------
__global__ __launch_bounds__(256) void edge_kernel(
    const float* __restrict__ x, const int* __restrict__ src,
    const int* __restrict__ dst, const float* __restrict__ edge_attr,
    const _Float16* __restrict__ Wef, const float* __restrict__ be,
    float* __restrict__ agg)
{
  const int tid = threadIdx.x;
  const int wave = tid >> 5, lane = tid & 31;
  const int halfsel = lane >> 4, lm = lane & 15;
  const int rowoff = halfsel * 8;
  const long e0 = ((long)blockIdx.x * 8 + wave) * 32;
  const long e1 = e0 + 16;

  // edge indices: 8 consecutive ints per tile -> two NT b128 loads each
  unsigned xb0[8], db0[8], xb1[8], db1[8];
  {
    v4i sa = __builtin_nontemporal_load((const v4i*)(src + e0 + rowoff));
    v4i sb = __builtin_nontemporal_load((const v4i*)(src + e0 + rowoff + 4));
    v4i da = __builtin_nontemporal_load((const v4i*)(dst + e0 + rowoff));
    v4i db = __builtin_nontemporal_load((const v4i*)(dst + e0 + rowoff + 4));
    v4i sc = __builtin_nontemporal_load((const v4i*)(src + e1 + rowoff));
    v4i sd = __builtin_nontemporal_load((const v4i*)(src + e1 + rowoff + 4));
    v4i dc = __builtin_nontemporal_load((const v4i*)(dst + e1 + rowoff));
    v4i dd = __builtin_nontemporal_load((const v4i*)(dst + e1 + rowoff + 4));
    #pragma unroll
    for (int r = 0; r < 4; ++r) {
      xb0[r] = (unsigned)sa[r] << 7;  xb0[r + 4] = (unsigned)sb[r] << 7;
      db0[r] = (unsigned)da[r] << 7;  db0[r + 4] = (unsigned)db[r] << 7;
      xb1[r] = (unsigned)sc[r] << 7;  xb1[r + 4] = (unsigned)sd[r] << 7;
      db1[r] = (unsigned)dc[r] << 7;  db1[r + 4] = (unsigned)dd[r] << 7;
    }
  }

  // A fragments (16x32 f16, ISA A layout), edge_attr streamed non-temporally
  v16h a0, a1;
  {
    const float* p0 = edge_attr + (e0 + lm) * EDD;
    const float* p1 = edge_attr + (e1 + lm) * EDD;
    const int o0 = halfsel * 8;
    v4f q0 = __builtin_nontemporal_load((const v4f*)(p0 + o0));
    v4f q1 = __builtin_nontemporal_load((const v4f*)(p0 + o0 + 4));
    v4f q2 = __builtin_nontemporal_load((const v4f*)(p0 + o0 + 16));
    v4f q3 = __builtin_nontemporal_load((const v4f*)(p0 + o0 + 20));
    v4f r0 = __builtin_nontemporal_load((const v4f*)(p1 + o0));
    v4f r1 = __builtin_nontemporal_load((const v4f*)(p1 + o0 + 4));
    v4f r2 = __builtin_nontemporal_load((const v4f*)(p1 + o0 + 16));
    v4f r3 = __builtin_nontemporal_load((const v4f*)(p1 + o0 + 20));
    #pragma unroll
    for (int j = 0; j < 4; ++j) {
      a0[j]      = (_Float16)q0[j];  a0[j + 4]  = (_Float16)q1[j];
      a0[j + 8]  = (_Float16)q2[j];  a0[j + 12] = (_Float16)q3[j];
      a1[j]      = (_Float16)r0[j];  a1[j + 4]  = (_Float16)r1[j];
      a1[j + 8]  = (_Float16)r2[j];  a1[j + 12] = (_Float16)r3[j];
    }
  }

  float bev[8];
  #pragma unroll
  for (int nt = 0; nt < 8; ++nt) bev[nt] = be[nt * 16 + lm];

  #pragma unroll
  for (int nt = 0; nt < 8; ++nt) {
    v16h b = *(const v16h*)(Wef + ((nt * 32 + lane) << 4));
    v8f acc0 = {}, acc1 = {};
    acc0 = __builtin_amdgcn_wmma_f32_16x16x32_f16(false, a0, false, b,
                                                  (short)0, acc0, false, false);
    acc1 = __builtin_amdgcn_wmma_f32_16x16x32_f16(false, a1, false, b,
                                                  (short)0, acc1, false, false);
    const unsigned n = nt * 16 + lm;

    // batch all 16 gathers for this tile before any atomic
    float xv0[8], xv1[8];
    #pragma unroll
    for (int r = 0; r < 8; ++r) xv0[r] = x[xb0[r] + n];
    #pragma unroll
    for (int r = 0; r < 8; ++r) xv1[r] = x[xb1[r] + n];

    const float bv = bev[nt];
    #pragma unroll
    for (int r = 0; r < 8; ++r) {
      float v = acc0[r] + bv + xv0[r];
      if (v > 0.0f) atomicAdd(&agg[db0[r] + n], v);   // relu: skip zeros
    }
    #pragma unroll
    for (int r = 0; r < 8; ++r) {
      float v = acc1[r] + bv + xv1[r];
      if (v > 0.0f) atomicAdd(&agg[db1[r] + n], v);
    }
  }
}

// ---------------------------------------------------------------------------
// Kernel 2: fused node MLP (no block barriers; waves fully independent).
//   h = x + agg;  h1 = silu(h@W1 + b1);  h2 = h1@W2 + b2
//   hwork = h2;   gsum[batch] += h2
// batch is sorted, so if the first and last row of a lane's D-slice share a
// graph id, all 8 rows do -> one locally-summed atomic instead of 8.
// ---------------------------------------------------------------------------
#define H1S 136   // padded row stride in halves; keeps 16B-aligned b128 loads
__global__ __launch_bounds__(256) void node_mlp_kernel(
    const float* __restrict__ x, const float* __restrict__ agg,
    const _Float16* __restrict__ W1f, const float* __restrict__ b1,
    const _Float16* __restrict__ W2f, const float* __restrict__ b2,
    const int* __restrict__ batch,
    float* __restrict__ hwork, float* __restrict__ gsum)
{
  __shared__ __align__(32) _Float16 h1s[8][16 * H1S];  // ~35 KB, per-wave slices

  const int tid = threadIdx.x;
  const int wave = tid >> 5, lane = tid & 31;
  const int halfsel = lane >> 4, lm = lane & 15;
  const int rowoff = halfsel * 8;
  const long nodebase = ((long)blockIdx.x * 8 + wave) * 16;
  const bool full = (nodebase + 16 <= NN);             // wave-uniform
  long arow = nodebase + lm; if (arow > NN - 1) arow = NN - 1;

  // A1 fragments: h = x + agg, 4 k-tiles
  v16h a1[4];
  {
    const float* px = x   + arow * HH;
    const float* pa = agg + arow * HH;
    #pragma unroll
    for (int kt = 0; kt < 4; ++kt) {
      const int o0 = kt * 32 + halfsel * 8;
      #pragma unroll
      for (int j = 0; j < 8; ++j) {
        a1[kt][j]     = (_Float16)(px[o0 + j]      + pa[o0 + j]);
        a1[kt][j + 8] = (_Float16)(px[o0 + 16 + j] + pa[o0 + 16 + j]);
      }
    }
  }

  float b1v[8], b2v[8];
  #pragma unroll
  for (int nt = 0; nt < 8; ++nt) {
    b1v[nt] = b1[nt * 16 + lm];
    b2v[nt] = b2[nt * 16 + lm];
  }

  // --- GEMM1 + SiLU -> per-wave h1 in LDS ---
  _Float16* myh1 = &h1s[wave][0];
  #pragma unroll
  for (int nt = 0; nt < 8; ++nt) {
    v8f acc = {};
    #pragma unroll
    for (int kt = 0; kt < 4; ++kt) {
      v16h b = *(const v16h*)(W1f + (((kt * 8 + nt) * 32 + lane) << 4));
      acc = __builtin_amdgcn_wmma_f32_16x16x32_f16(false, a1[kt], false, b,
                                                   (short)0, acc, false, false);
    }
    const int n = nt * 16 + lm;
    #pragma unroll
    for (int r = 0; r < 8; ++r) {
      float v = acc[r] + b1v[nt];
      v = v / (1.0f + __expf(-v));                 // silu
      myh1[(rowoff + r) * H1S + n] = (_Float16)v;
    }
  }

  // A2 fragments from LDS (same-wave DS ordering guarantees visibility)
  v16h a2[4];
  #pragma unroll
  for (int kt = 0; kt < 4; ++kt) {
    const int o0 = kt * 32 + halfsel * 8;
    const _Float16* ph = myh1 + lm * H1S;
    v8h lo = *(const v8h*)(ph + o0);
    v8h hi = *(const v8h*)(ph + o0 + 16);
    #pragma unroll
    for (int j = 0; j < 8; ++j) { a2[kt][j] = lo[j]; a2[kt][j + 8] = hi[j]; }
  }

  // batch ids for this lane's D rows
  int bi[8];
  #pragma unroll
  for (int r = 0; r < 8; ++r) {
    long nd = nodebase + rowoff + r;
    bi[r] = batch[nd > NN - 1 ? NN - 1 : nd];
  }
  const bool onegraph = (bi[0] == bi[7]);  // sorted batch => all equal

  // --- GEMM2 + epilogue ---
  #pragma unroll
  for (int nt = 0; nt < 8; ++nt) {
    v8f acc = {};
    #pragma unroll
    for (int kt = 0; kt < 4; ++kt) {
      v16h b = *(const v16h*)(W2f + (((kt * 8 + nt) * 32 + lane) << 4));
      acc = __builtin_amdgcn_wmma_f32_16x16x32_f16(false, a2[kt], false, b,
                                                   (short)0, acc, false, false);
    }
    const unsigned n = nt * 16 + lm;
    const unsigned rowb = (unsigned)(nodebase + rowoff) << 7;
    if (full) {                                      // scalar-branch fast path
      float v[8];
      #pragma unroll
      for (int r = 0; r < 8; ++r) {
        v[r] = acc[r] + b2v[nt];
        hwork[rowb + ((unsigned)r << 7) + n] = v[r];
      }
      if (onegraph) {
        float s = ((v[0] + v[1]) + (v[2] + v[3])) +
                  ((v[4] + v[5]) + (v[6] + v[7]));
        atomicAdd(&gsum[((unsigned)bi[0] << 7) + n], s);
      } else {
        #pragma unroll
        for (int r = 0; r < 8; ++r)
          atomicAdd(&gsum[((unsigned)bi[r] << 7) + n], v[r]);
      }
    } else {
      #pragma unroll
      for (int r = 0; r < 8; ++r) {
        if (nodebase + rowoff + r < NN) {
          float v = acc[r] + b2v[nt];
          hwork[rowb + ((unsigned)r << 7) + n] = v;
          atomicAdd(&gsum[((unsigned)bi[r] << 7) + n], v);
        }
      }
    }
  }
}

// ---------------------------------------------------------------------------
// GraphNorm / FiLM kernels
// ---------------------------------------------------------------------------
__global__ void count_kernel(const int* __restrict__ batch, float* __restrict__ cnt) {
  int i = blockIdx.x * blockDim.x + threadIdx.x;
  if (i < NN) atomicAdd(&cnt[batch[i]], 1.0f);
}

__global__ void mean_kernel(const float* __restrict__ gsum,
                            const float* __restrict__ cnt,
                            float* __restrict__ meanb) {
  int i = blockIdx.x * blockDim.x + threadIdx.x;
  if (i < GG * HH) {
    float c = cnt[i >> 7]; c = c < 1.0f ? 1.0f : c;
    meanb[i] = gsum[i] / c;
  }
}

// run-accumulating variance reduction: each thread walks CN consecutive nodes
// of one column; sorted batch means one atomic per graph-run (~32x fewer).
#define CN 32
__global__ __launch_bounds__(256) void center_kernel(
    const float* __restrict__ meanb, const float* __restrict__ ms,
    const int* __restrict__ batch, const float* __restrict__ hwork,
    float* __restrict__ vsum)
{
  const int tid = threadIdx.x;
  const unsigned col = tid & 127;
  const unsigned chunk = blockIdx.x * 2 + (tid >> 7);
  unsigned n0 = chunk * CN;
  if (n0 >= NN) return;
  unsigned nend = n0 + CN; if (nend > NN) nend = NN;
  const float msc = ms[col];
  int g = batch[n0];
  float mrun = meanb[((unsigned)g << 7) + col] * msc;
  float acc = 0.0f;
  for (unsigned node = n0; node < nend; ++node) {
    int gg = batch[node];
    if (gg != g) {
      atomicAdd(&vsum[((unsigned)g << 7) + col], acc);
      acc = 0.0f; g = gg;
      mrun = meanb[((unsigned)g << 7) + col] * msc;
    }
    float o = hwork[(node << 7) + col] - mrun;
    acc += o * o;
  }
  atomicAdd(&vsum[((unsigned)g << 7) + col], acc);
}

__global__ void vinv_kernel(const float* __restrict__ vsum,
                            const float* __restrict__ cnt,
                            float* __restrict__ vinv) {
  int i = blockIdx.x * blockDim.x + threadIdx.x;
  if (i < GG * HH) {
    float c = cnt[i >> 7]; c = c < 1.0f ? 1.0f : c;
    vinv[i] = rsqrtf(vsum[i] / c + EPS_GN);
  }
}

__global__ void film_kernel(const float* __restrict__ te,
                            const float* __restrict__ Wg, const float* __restrict__ bg,
                            const float* __restrict__ Wb, const float* __restrict__ bbv,
                            float* __restrict__ gamma, float* __restrict__ beta) {
  int i = blockIdx.x * blockDim.x + threadIdx.x;
  if (i < GG * HH) {
    const int g = i >> 7, h = i & 127;
    const float* t = te + (long)g * TDD;
    float sg = 0.0f, sb = 0.0f;
    for (int k = 0; k < TDD; ++k) {
      const float tv = t[k];
      sg += tv * Wg[k * HH + h];
      sb += tv * Wb[k * HH + h];
    }
    gamma[i] = sg + bg[h] + 1.0f;
    beta[i]  = sb + bbv[h];
  }
}

// float4-vectorized final pass; recomputes centering from L2-resident h2
__global__ __launch_bounds__(256) void final_kernel(
    const float* __restrict__ x, const float* __restrict__ hwork,
    const float* __restrict__ meanb, const float* __restrict__ ms,
    const float* __restrict__ vinv,
    const float* __restrict__ gw, const float* __restrict__ gb,
    const float* __restrict__ gamma, const float* __restrict__ beta,
    const int* __restrict__ batch, float* __restrict__ out)
{
  unsigned q = blockIdx.x * blockDim.x + threadIdx.x;   // one float4 per thread
  if (q < (unsigned)(NN * HH / 4)) {
    const unsigned node = q >> 5;          // 32 quads per node row
    const unsigned h4 = (q & 31) << 2;
    const unsigned g = (unsigned)batch[node];
    const unsigned gb4 = (g << 7) + h4;
    v4f hv  = *(const v4f*)(hwork + (q << 2));
    v4f xv  = *(const v4f*)(x + (q << 2));
    v4f mb  = *(const v4f*)(meanb + gb4);
    v4f msv = *(const v4f*)(ms + h4);
    v4f vi  = *(const v4f*)(vinv + gb4);
    v4f gwv = *(const v4f*)(gw + h4);
    v4f gbv = *(const v4f*)(gb + h4);
    v4f gav = *(const v4f*)(gamma + gb4);
    v4f bev = *(const v4f*)(beta + gb4);
    v4f o;
    #pragma unroll
    for (int j = 0; j < 4; ++j) {
      float v = (hv[j] - mb[j] * msv[j]) * vi[j];
      v = gwv[j] * v + gbv[j];
      v = gav[j] * v + bev[j];
      v = v / (1.0f + __expf(-v));   // silu
      o[j] = xv[j] + v;
    }
    *(v4f*)(out + (q << 2)) = o;
  }
}

// ---------------------------------------------------------------------------
extern "C" void kernel_launch(void* const* d_in, const int* in_sizes, int n_in,
                              void* d_out, int out_size, void* d_ws, size_t ws_size,
                              hipStream_t stream) {
  (void)in_sizes; (void)n_in; (void)out_size; (void)ws_size;
  const float* x    = (const float*)d_in[0];
  const int*   ei   = (const int*)  d_in[1];   // [2,E]: src=ei[0:E), dst=ei[E:2E)
  const float* ea   = (const float*)d_in[2];
  const int*   bat  = (const int*)  d_in[3];
  const float* te   = (const float*)d_in[4];
  const float* We   = (const float*)d_in[5];
  const float* be   = (const float*)d_in[6];
  const float* W1   = (const float*)d_in[7];
  const float* b1   = (const float*)d_in[8];
  const float* W2   = (const float*)d_in[9];
  const float* b2   = (const float*)d_in[10];
  const float* gnw  = (const float*)d_in[11];
  const float* gnb  = (const float*)d_in[12];
  const float* gnms = (const float*)d_in[13];
  const float* Wg   = (const float*)d_in[14];
  const float* bg   = (const float*)d_in[15];
  const float* Wb   = (const float*)d_in[16];
  const float* bb   = (const float*)d_in[17];
  float* out = (float*)d_out;

  // workspace layout (all offsets multiples of 256B)
  char* ws = (char*)d_ws;
  const size_t SZ_NH = (size_t)NN * HH * sizeof(float);   // 51.2 MB
  const size_t SZ_GH = (size_t)GG * HH * sizeof(float);   // 256 KB
  size_t o = 0;
  float* agg   = (float*)(ws + o);  o += SZ_NH;
  float* hwork = (float*)(ws + o);  o += SZ_NH;
  float* gsum  = (float*)(ws + o);  o += SZ_GH;
  float* vsum  = (float*)(ws + o);  o += SZ_GH;
  float* meanb = (float*)(ws + o);  o += SZ_GH;
  float* vinv  = (float*)(ws + o);  o += SZ_GH;
  float* gamma = (float*)(ws + o);  o += SZ_GH;
  float* beta  = (float*)(ws + o);  o += SZ_GH;
  float* cnt   = (float*)(ws + o);  o += 4096;            // G floats, padded
  _Float16* Wef = (_Float16*)(ws + o); o += 8 * 32 * 16 * 2;     // 8 KB
  _Float16* W1f = (_Float16*)(ws + o); o += 32 * 32 * 16 * 2;    // 32 KB
  _Float16* W2f = (_Float16*)(ws + o); o += 32 * 32 * 16 * 2;    // 32 KB

  // zero accumulators (stream-ordered; graph-capturable)
  hipMemsetAsync(agg,  0, SZ_NH, stream);
  hipMemsetAsync(gsum, 0, SZ_GH, stream);
  hipMemsetAsync(vsum, 0, SZ_GH, stream);
  hipMemsetAsync(cnt,  0, GG * sizeof(float), stream);

  // 0. one-time weight fragment prep (f32 -> f16 B-operand layout)
  prep_frag_kernel<<<1, 256, 0, stream>>>(We, Wef, 1);
  prep_frag_kernel<<<4, 256, 0, stream>>>(W1, W1f, 4);
  prep_frag_kernel<<<4, 256, 0, stream>>>(W2, W2f, 4);

  // 1. edge stage: 256 edges/block (8 waves x 32 edges)
  edge_kernel<<<EE / 256, 256, 0, stream>>>(x, ei, ei + EE, ea, Wef, be, agg);

  // 2. node MLP: 128 nodes/block (8 waves x 16 nodes)
  node_mlp_kernel<<<(NN + 127) / 128, 256, 0, stream>>>(x, agg, W1f, b1, W2f, b2,
                                                        bat, hwork, gsum);

  // 3. GraphNorm + FiLM
  count_kernel <<<(NN + 255) / 256, 256, 0, stream>>>(bat, cnt);
  mean_kernel  <<<(GG * HH + 255) / 256, 256, 0, stream>>>(gsum, cnt, meanb);
  {
    const int chunks = (NN + CN - 1) / CN;          // 3125 node-chunks
    center_kernel<<<(chunks + 1) / 2, 256, 0, stream>>>(meanb, gnms, bat,
                                                        hwork, vsum);
  }
  vinv_kernel  <<<(GG * HH + 255) / 256, 256, 0, stream>>>(vsum, cnt, vinv);
  film_kernel  <<<(GG * HH + 255) / 256, 256, 0, stream>>>(te, Wg, bg, Wb, bb,
                                                           gamma, beta);
  final_kernel <<<(NN * HH / 4 + 255) / 256, 256, 0, stream>>>(x, hwork, meanb,
                                                               gnms, vinv, gnw,
                                                               gnb, gamma, beta,
                                                               bat, out);
}
